// MambaBlock_84980222919164
// MI455X (gfx1250) — matrix operand order repaired and checked by reading, not compiled, not run on hardware
//
#include <hip/hip_runtime.h>
#include <hip/hip_bf16.h>
#include <math.h>

#define D_MODEL 1024
#define D_INNER 2048
#define D_CONV  4
#define NB      4
#define NT      4096
#define NROWS   (NB * NT)        // 16384
#define LN_EPS  1e-5f

typedef unsigned int u32;
typedef __attribute__((ext_vector_type(16))) __bf16 v16bf;
typedef __attribute__((ext_vector_type(8)))  __bf16 v8bf;
typedef __attribute__((ext_vector_type(8)))  float  v8f;
typedef __attribute__((ext_vector_type(4)))  u32    v4u;
typedef __attribute__((ext_vector_type(8)))  int    v8i;
typedef __attribute__((ext_vector_type(4)))  int    v4i;

// LDS tile geometry: K-chunk = 32 bf16 per row = 64B, TDM pads 4 DWORDs per
// 16 DWORDs stored -> 80B row pitch (conflict-free ds_load_b128: bank step 20)
#define ROW_B    80
#define A_ROWS   64
#define B_ROWS   256
#define A_STAGE  (A_ROWS * ROW_B)   // 5120 B
#define B_STAGE  (B_ROWS * ROW_B)   // 20480 B

// ---------------------------------------------------------------------------
// LDS byte offset of a generic pointer (addrspacecast to LDS)
// ---------------------------------------------------------------------------
typedef __attribute__((address_space(3))) char lds_char;
__device__ __forceinline__ u32 lds_off_of(const void* p) {
  return (u32)(uintptr_t)(lds_char*)(void*)p;
}

// ---------------------------------------------------------------------------
// Tensor Data Mover: 2-D bf16 tile (32 K-elems x `rows`) global -> LDS.
// D# per CDNA5 ISA; pad 4 DWORDs every 16 DWORDs.
// 6-arg builtin form: (g0, g1, g2, g3, g_extra, cpol)
// ---------------------------------------------------------------------------
__device__ __forceinline__ void tdm_load_2d(u32 lds_byte_off,
                                            const __bf16* gptr, u32 rows,
                                            u32 stride_elems) {
  const unsigned long long ga = (unsigned long long)(uintptr_t)gptr;
  v4u g0;
  g0[0] = 1u;                                   // count=1, user descriptor
  g0[1] = lds_byte_off;                         // lds_addr
  g0[2] = (u32)ga;                              // global_addr[31:0]
  g0[3] = (u32)((ga >> 32) & 0x01ffffffu)       // global_addr[56:32]
          | (2u << 30);                         // type = 2 (image)
  v8i g1;
  g1[0] = (int)((1u << 16)      // data_size = 1 -> 2 bytes
                | (1u << 20)    // pad_enable
                | (3u << 22)    // pad_interval code 3 = 16 DWORDs (64B row)
                | (3u << 25));  // pad_amount  code 3 = 4 DWORDs (16B)
  g1[1] = (int)((32u & 0xffffu) << 16);                    // tensor_dim0 lo16
  g1[2] = (int)(((32u >> 16) & 0xffffu) | ((rows & 0xffffu) << 16));
  g1[3] = (int)(((rows >> 16) & 0xffffu) | (32u << 16));   // tile_dim0 = 32
  g1[4] = (int)(rows & 0xffffu);                           // tile_dim1 = rows
  g1[5] = (int)stride_elems;                               // dim0_stride lo32
  g1[6] = 0;
  g1[7] = 0;
  v4i gz4 = {0, 0, 0, 0};
  v8i gz8 = {0, 0, 0, 0, 0, 0, 0, 0};
  __builtin_amdgcn_tensor_load_to_lds(g0, g1, gz4, gz4, gz8, 0);
}

// ---------------------------------------------------------------------------
// WMMA fragment loads from padded LDS tiles (CDNA5 ISA 7.12.2 layouts):
//   A 16x32 bf16: lane%16 = M row; lane<16 -> K {0..7,16..23}, else {8..15,24..31}
//   B 32x16 bf16: lane%16 = N row(col); lane<16 -> K {0..7,16..23}, else {8..15,24..31}
// ---------------------------------------------------------------------------
__device__ __forceinline__ v16bf pack16(v8bf lo, v8bf hi) {
  v16bf r;
#pragma unroll
  for (int i = 0; i < 8; ++i) { r[i] = lo[i]; r[8 + i] = hi[i]; }
  return r;
}
__device__ __forceinline__ v16bf lds_afrag(const char* rowp, int half) {
  return pack16(*(const v8bf*)(rowp + half * 16),
                *(const v8bf*)(rowp + 32 + half * 16));
}
__device__ __forceinline__ v16bf lds_bfrag(const char* rowp, int half) {
  return pack16(*(const v8bf*)(rowp + half * 16),
                *(const v8bf*)(rowp + 32 + half * 16));
}

#define WMMA_BF16(A, Bf, C)                                                   \
  __builtin_amdgcn_wmma_f32_16x16x32_bf16(false, (A), false, (Bf), (short)0,  \
                                          (C), false, false)

// ---------------------------------------------------------------------------
// TDM-staged block GEMM body.  Block = 256 threads (8 waves) computing a
// 64(M) x 256(N) tile; wave (wm,wn) owns a 32x64 sub-tile.  K-chunks of 32
// are DMA'd (double-buffered) into LDS by wave 0 while the other stage feeds
// the 8-WMMA bursts.  Accumulators are named locals -> stable loop registers.
// ---------------------------------------------------------------------------
__device__ __forceinline__ void gemm_staged(const __bf16* __restrict__ Ag,
                                            int lda,
                                            const __bf16* __restrict__ Bg,
                                            int ldb, int K, v8f acc[8]) {
  __shared__ __align__(128) char lds_a[2 * A_STAGE];
  __shared__ __align__(128) char lds_b[2 * B_STAGE];

  const int tid  = threadIdx.x;
  const int wave = tid >> 5;
  const int lane = tid & 31;
  const int half = lane >> 4;
  const int l16  = lane & 15;
  const int wm   = wave >> 2;   // 0..1  -> M sub-block of 32
  const int wn   = wave & 3;    // 0..3  -> N sub-block of 64

  v8f c0 = {}, c1 = {}, c2 = {}, c3 = {};
  v8f c4 = {}, c5 = {}, c6 = {}, c7 = {};

  const int nsteps = K >> 5;
  if (wave == 0) {
    tdm_load_2d(lds_off_of(lds_a), Ag, A_ROWS, (u32)lda);
    tdm_load_2d(lds_off_of(lds_b), Bg, B_ROWS, (u32)ldb);
  }

  for (int i = 0; i < nsteps; ++i) {
    const int s = i & 1;
    if (i + 1 < nsteps) {
      if (wave == 0) {
        const int kn = (i + 1) << 5;
        tdm_load_2d(lds_off_of(lds_a) + (u32)((s ^ 1) * A_STAGE), Ag + kn,
                    A_ROWS, (u32)lda);
        tdm_load_2d(lds_off_of(lds_b) + (u32)((s ^ 1) * B_STAGE), Bg + kn,
                    B_ROWS, (u32)ldb);
        __builtin_amdgcn_s_wait_tensorcnt(2);  // 2 oldest (stage i) complete
      }
    } else if (wave == 0) {
      __builtin_amdgcn_s_wait_tensorcnt(0);
    }
    __syncthreads();  // stage i visible to all waves

    const char* Ab = lds_a + s * A_STAGE + (size_t)(wm * 32) * ROW_B;
    const char* Bb = lds_b + s * B_STAGE + (size_t)(wn * 64) * ROW_B;

    const v16bf a0 = lds_afrag(Ab + (size_t)l16 * ROW_B, half);
    const v16bf a1 = lds_afrag(Ab + (size_t)(16 + l16) * ROW_B, half);
    const v16bf b0 = lds_bfrag(Bb + (size_t)(0 * 16 + l16) * ROW_B, half);
    const v16bf b1 = lds_bfrag(Bb + (size_t)(1 * 16 + l16) * ROW_B, half);
    const v16bf b2 = lds_bfrag(Bb + (size_t)(2 * 16 + l16) * ROW_B, half);
    const v16bf b3 = lds_bfrag(Bb + (size_t)(3 * 16 + l16) * ROW_B, half);

    c0 = WMMA_BF16(a0, b0, c0);
    c4 = WMMA_BF16(a1, b0, c4);
    c1 = WMMA_BF16(a0, b1, c1);
    c5 = WMMA_BF16(a1, b1, c5);
    c2 = WMMA_BF16(a0, b2, c2);
    c6 = WMMA_BF16(a1, b2, c6);
    c3 = WMMA_BF16(a0, b3, c3);
    c7 = WMMA_BF16(a1, b3, c7);

    __syncthreads();  // all reads of stage i done before it is overwritten
  }

  acc[0] = c0; acc[1] = c1; acc[2] = c2; acc[3] = c3;
  acc[4] = c4; acc[5] = c5; acc[6] = c6; acc[7] = c7;
}

// ---------------------------------------------------------------------------
// LayerNorm (f32 in) -> bf16 out.  One 256-thread block per row.
// ---------------------------------------------------------------------------
__global__ void ln_bf16_kernel(const float* __restrict__ x,
                               const float* __restrict__ gamma,
                               const float* __restrict__ beta,
                               __bf16* __restrict__ xn) {
  const int row = blockIdx.x;
  const float* xr = x + (size_t)row * D_MODEL;

  float v[4];
  float s1 = 0.f, s2 = 0.f;
#pragma unroll
  for (int i = 0; i < 4; ++i) {
    v[i] = xr[threadIdx.x + i * 256];
    s1 += v[i];
    s2 += v[i] * v[i];
  }
#pragma unroll
  for (int off = 16; off > 0; off >>= 1) {
    s1 += __shfl_down(s1, off, 32);
    s2 += __shfl_down(s2, off, 32);
  }
  __shared__ float red1[8], red2[8];
  __shared__ float smu, srstd;
  const int lane = threadIdx.x & 31;
  const int wv   = threadIdx.x >> 5;
  if (lane == 0) { red1[wv] = s1; red2[wv] = s2; }
  __syncthreads();
  if (threadIdx.x == 0) {
    float a = 0.f, b = 0.f;
#pragma unroll
    for (int i = 0; i < 8; ++i) { a += red1[i]; b += red2[i]; }
    const float mu  = a * (1.0f / D_MODEL);
    const float var = b * (1.0f / D_MODEL) - mu * mu;
    smu = mu;
    srstd = rsqrtf(var + LN_EPS);
  }
  __syncthreads();
  const float mu = smu, rstd = srstd;
  __bf16* xo = xn + (size_t)row * D_MODEL;
#pragma unroll
  for (int i = 0; i < 4; ++i) {
    const int c = threadIdx.x + i * 256;
    xo[c] = (__bf16)(((v[i] - mu) * rstd) * gamma[c] + beta[c]);
  }
}

// ---------------------------------------------------------------------------
// f32 -> bf16 convert
// ---------------------------------------------------------------------------
__global__ void f32_to_bf16_kernel(const float* __restrict__ in,
                                   __bf16* __restrict__ out, int n) {
  const int i = blockIdx.x * 256 + threadIdx.x;
  if (i < n) out[i] = (__bf16)in[i];
}

// ---------------------------------------------------------------------------
// GEMM1: xz[16384,4096] = xn[16384,1024] x W_in[4096,1024]^T  (bf16 out)
// grid: (16384/64, 4096/256)
// ---------------------------------------------------------------------------
__global__ __launch_bounds__(256, 1) void gemm_in_kernel(
    const __bf16* __restrict__ xn, const __bf16* __restrict__ Wi,
    __bf16* __restrict__ xz) {
  const int wave = threadIdx.x >> 5;
  const int wm = wave >> 2, wn = wave & 3;
  const int mblk = blockIdx.x * 64;
  const int nblk = blockIdx.y * 256;

  v8f acc[8];
  gemm_staged(xn + (size_t)mblk * D_MODEL, D_MODEL,
              Wi + (size_t)nblk * D_MODEL, D_MODEL, D_MODEL, acc);

  const int lane = threadIdx.x & 31;
  const int half = lane >> 4;
  const int l16  = lane & 15;
  const int m0w = mblk + wm * 32;
  const int n0w = nblk + wn * 64;
#pragma unroll
  for (int g = 0; g < 2; ++g) {
#pragma unroll
    for (int j = 0; j < 4; ++j) {
#pragma unroll
      for (int r = 0; r < 8; ++r) {
        const int m = m0w + g * 16 + r + half * 8;
        const int n = n0w + j * 16 + l16;
        xz[(size_t)m * (2 * D_INNER) + n] = (__bf16)acc[g * 4 + j][r];
      }
    }
  }
}

// ---------------------------------------------------------------------------
// Causal depthwise conv (4 taps) + SiLU(x_conv)*SiLU(z) gate.  bf16 out.
// ---------------------------------------------------------------------------
__device__ __forceinline__ float silu_f(float v) {
  return v / (1.0f + __expf(-v));
}

__global__ void conv_gate_kernel(const __bf16* __restrict__ xz,
                                 const float* __restrict__ conv_w,
                                 const float* __restrict__ conv_b,
                                 __bf16* __restrict__ y) {
  const int idx = blockIdx.x * 256 + threadIdx.x;
  if (idx >= NROWS * D_INNER) return;
  const int c   = idx & (D_INNER - 1);
  const int row = idx >> 11;
  const int t   = row & (NT - 1);

  float acc = conv_b[c];
#pragma unroll
  for (int j = 0; j < D_CONV; ++j) {
    const int tt = t - (D_CONV - 1) + j;
    if (tt >= 0) {
      acc += conv_w[c * D_CONV + j] *
             (float)xz[(size_t)(row - (D_CONV - 1) + j) * (2 * D_INNER) + c];
    }
  }
  const float z = (float)xz[(size_t)row * (2 * D_INNER) + D_INNER + c];
  y[idx] = (__bf16)(silu_f(acc) * silu_f(z));
}

// ---------------------------------------------------------------------------
// GEMM2: out[16384,1024] = residual + y[16384,2048] x W_out[1024,2048]^T
// grid: (16384/64, 1024/256)
// ---------------------------------------------------------------------------
__global__ __launch_bounds__(256, 1) void gemm_out_kernel(
    const __bf16* __restrict__ y, const __bf16* __restrict__ Wo,
    const float* __restrict__ resid, float* __restrict__ out) {
  const int wave = threadIdx.x >> 5;
  const int wm = wave >> 2, wn = wave & 3;
  const int mblk = blockIdx.x * 64;
  const int nblk = blockIdx.y * 256;

  v8f acc[8];
  gemm_staged(y + (size_t)mblk * D_INNER, D_INNER,
              Wo + (size_t)nblk * D_INNER, D_INNER, D_INNER, acc);

  const int lane = threadIdx.x & 31;
  const int half = lane >> 4;
  const int l16  = lane & 15;
  const int m0w = mblk + wm * 32;
  const int n0w = nblk + wn * 64;
#pragma unroll
  for (int g = 0; g < 2; ++g) {
#pragma unroll
    for (int j = 0; j < 4; ++j) {
#pragma unroll
      for (int r = 0; r < 8; ++r) {
        const int m = m0w + g * 16 + r + half * 8;
        const int n = n0w + j * 16 + l16;
        out[(size_t)m * D_MODEL + n] =
            resid[(size_t)m * D_MODEL + n] + acc[g * 4 + j][r];
      }
    }
  }
}

// ---------------------------------------------------------------------------
// Launch
// ---------------------------------------------------------------------------
extern "C" void kernel_launch(void* const* d_in, const int* in_sizes, int n_in,
                              void* d_out, int out_size, void* d_ws,
                              size_t ws_size, hipStream_t stream) {
  const float* x      = (const float*)d_in[0];
  const float* gamma  = (const float*)d_in[1];
  const float* beta   = (const float*)d_in[2];
  const float* W_in   = (const float*)d_in[3];
  const float* conv_w = (const float*)d_in[4];
  const float* conv_b = (const float*)d_in[5];
  const float* W_out  = (const float*)d_in[6];
  float* out = (float*)d_out;

  char* ws = (char*)d_ws;
  __bf16* xn = (__bf16*)ws;  ws += (size_t)NROWS * D_MODEL * 2;
  __bf16* Wi = (__bf16*)ws;  ws += (size_t)(2 * D_INNER) * D_MODEL * 2;
  __bf16* Wo = (__bf16*)ws;  ws += (size_t)D_MODEL * D_INNER * 2;
  __bf16* xz = (__bf16*)ws;  ws += (size_t)NROWS * (2 * D_INNER) * 2;
  __bf16* yb = (__bf16*)ws;  ws += (size_t)NROWS * D_INNER * 2;

  ln_bf16_kernel<<<NROWS, 256, 0, stream>>>(x, gamma, beta, xn);

  {
    const int n1 = 2 * D_INNER * D_MODEL;
    f32_to_bf16_kernel<<<(n1 + 255) / 256, 256, 0, stream>>>(W_in, Wi, n1);
    const int n2 = D_MODEL * D_INNER;
    f32_to_bf16_kernel<<<(n2 + 255) / 256, 256, 0, stream>>>(W_out, Wo, n2);
  }

  {
    dim3 grid(NROWS / 64, (2 * D_INNER) / 256);
    gemm_in_kernel<<<grid, 256, 0, stream>>>(xn, Wi, xz);
  }

  {
    const int n = NROWS * D_INNER;
    conv_gate_kernel<<<(n + 255) / 256, 256, 0, stream>>>(xz, conv_w, conv_b,
                                                          yb);
  }

  {
    dim3 grid(NROWS / 64, D_MODEL / 256);
    gemm_out_kernel<<<grid, 256, 0, stream>>>(yb, Wo, x, out);
  }
}
